// Backbone1_62947040690721
// MI455X (gfx1250) — compile-verified, matrix operand order checked by reading, not run on hardware
//
#include <hip/hip_runtime.h>
#include <cmath>

#define B_ 16
#define T_ 512
#define NV 21
#define PS 16
#define STRIDE_ 8
#define PRED 96
#define DM 128
#define DS 16
#define DC 4
#define DIN 256
#define DTR 8
#define SEA 512
#define PN 64
#define BN (B_*NV)        /* 336 */
#define ROWS (BN*PN)      /* 21504 */
#define EPS 1e-5f

typedef __attribute__((ext_vector_type(16))) _Float16 v16h;
typedef __attribute__((ext_vector_type(8)))  float    v8f;

// ---------------------------------------------------------------------------
// helpers
// ---------------------------------------------------------------------------
__device__ __forceinline__ int wmma_k_of(int lane, int h) {
  // 16-bit A/B fragment k-mapping for 16x16x32 (wave32):
  // lanes 0-15 hold K 0-7 (v0-3) and 16-23 (v4-7); lanes 16-31 hold K 8-15 / 24-31
  int base = (lane >> 4) << 3;           // 0 or 8
  int v = h >> 1, p = h & 1;
  return (v < 4) ? (base + 2 * v + p) : (16 + base + 2 * (v - 4) + p);
}
__device__ __forceinline__ float gelu_exact(float x) {
  return 0.5f * x * (1.0f + erff(x * 0.70710678f));
}
__device__ __forceinline__ float silu_f(float x) { return x / (1.0f + __expf(-x)); }
__device__ __forceinline__ float softplus_f(float x) {
  return (x > 20.0f) ? x : log1pf(__expf(x));
}

// ---------------------------------------------------------------------------
// 1. RevIN statistics: mean/std over T per (b,v)
// ---------------------------------------------------------------------------
__global__ void stats_kernel(const float* __restrict__ x, float* __restrict__ mean,
                             float* __restrict__ stdv) {
  int bv = blockIdx.x; int b = bv / NV, v = bv % NV;
  __shared__ float s1[256], s2[256];
  float a0 = x[(b * T_ + threadIdx.x) * NV + v];
  float a1 = x[(b * T_ + threadIdx.x + 256) * NV + v];
  s1[threadIdx.x] = a0 + a1;
  s2[threadIdx.x] = a0 * a0 + a1 * a1;
  __syncthreads();
  for (int o = 128; o > 0; o >>= 1) {
    if ((int)threadIdx.x < o) { s1[threadIdx.x] += s1[threadIdx.x + o]; s2[threadIdx.x] += s2[threadIdx.x + o]; }
    __syncthreads();
  }
  if (threadIdx.x == 0) {
    float m = s1[0] / (float)T_;
    float var = s2[0] / (float)T_ - m * m;
    mean[bv] = m;
    stdv[bv] = sqrtf(var + EPS);
  }
}

// ---------------------------------------------------------------------------
// 2. normalize + patchify + mlp1 (16->128) -> h32 (residual) + hA fragments
// ---------------------------------------------------------------------------
__global__ void patch_mlp1_kernel(const float* __restrict__ x, const float* __restrict__ revin_w,
                                  const float* __restrict__ revin_b, const float* __restrict__ w1,
                                  const float* __restrict__ b1, const float* __restrict__ mean,
                                  const float* __restrict__ stdv, float* __restrict__ h32,
                                  _Float16* __restrict__ hA) {
  int seq = blockIdx.x; int b = seq / NV, v = seq % NV;
  __shared__ float patch[PN][PS];     // 4 KB
  __shared__ float hS[PN][DM];        // 32 KB
  float m = mean[seq], sd = stdv[seq];
  float rw = revin_w[v], rb = revin_b[v];
  for (int e = threadIdx.x; e < PN * PS; e += blockDim.x) {
    int p = e / PS, i = e % PS;
    int t = p * STRIDE_ + i; if (t > T_ - 1) t = T_ - 1;   // right-pad with last value
    patch[p][i] = (x[(b * T_ + t) * NV + v] - m) / sd * rw + rb;
  }
  __syncthreads();
  for (int o = threadIdx.x; o < PN * DM; o += blockDim.x) {
    int l = o / DM, d = o % DM;
    float acc = b1[d];
#pragma unroll
    for (int i = 0; i < PS; i++) acc += patch[l][i] * w1[d * PS + i];
    hS[l][d] = acc;
    h32[((size_t)(seq * PN + l)) * DM + d] = acc;
  }
  __syncthreads();
  // emit A fragments for this 64x128 tile block (4 mtiles x 4 ktiles)
  for (int pp = threadIdx.x; pp < 16 * 32; pp += blockDim.x) {
    int lt = pp >> 5, lane = pp & 31;
    int lmt = lt >> 2, lkt = lt & 3;
    int lr = lmt * 16 + (lane & 15);
    v16h tmp;
#pragma unroll
    for (int h = 0; h < 16; h++) tmp[h] = (_Float16)hS[lr][lkt * 32 + wmma_k_of(lane, h)];
    size_t tile = (size_t)(seq * 4 + lmt) * 4 + lkt;       // Ktiles = 4
    *(v16h*)(hA + tile * 512 + lane * 16) = tmp;
  }
}

// ---------------------------------------------------------------------------
// fragment swizzles (f32 row-major -> f16 WMMA fragment order)
// ---------------------------------------------------------------------------
__global__ void swizzleA_kernel(const float* __restrict__ X, _Float16* __restrict__ out,
                                int Mtiles, int Ktiles, int K) {
  int t = blockIdx.x * blockDim.x + threadIdx.x;
  if (t >= Mtiles * Ktiles * 32) return;
  int lane = t & 31; int tile = t >> 5;
  int mt = tile / Ktiles, kt = tile % Ktiles;
  int mrow = mt * 16 + (lane & 15);
  v16h tmp;
#pragma unroll
  for (int h = 0; h < 16; h++)
    tmp[h] = (_Float16)X[(size_t)mrow * K + kt * 32 + wmma_k_of(lane, h)];
  *(v16h*)(out + ((size_t)tile * 512 + lane * 16)) = tmp;
}

__global__ void swizzleB_kernel(const float* __restrict__ W, _Float16* __restrict__ out,
                                int Ntiles, int Ktiles, int K) {
  // W is N x K row-major; logical B = W^T (K x N)
  int t = blockIdx.x * blockDim.x + threadIdx.x;
  if (t >= Ntiles * Ktiles * 32) return;
  int lane = t & 31; int tile = t >> 5;
  int nt = tile / Ktiles, kt = tile % Ktiles;
  int n = nt * 16 + (lane & 15);
  v16h tmp;
#pragma unroll
  for (int h = 0; h < 16; h++)
    tmp[h] = (_Float16)W[(size_t)n * K + kt * 32 + wmma_k_of(lane, h)];
  *(v16h*)(out + ((size_t)tile * 512 + lane * 16)) = tmp;
}

// ---------------------------------------------------------------------------
// M-blocked WMMA GEMM, ping-pong double-buffered over K (Ktiles must be even,
// >= 2): no register rotation copies; each load set has a full 4-WMMA compute
// stage of latency cover.  out(M x ldo) = A(MxK) * B(KxN) [+bias][gelu]
// ---------------------------------------------------------------------------
template <int MBLK>
__global__ void wmma_gemm_kernel(const _Float16* __restrict__ A, const _Float16* __restrict__ Bf,
                                 const float* __restrict__ bias, float* __restrict__ out,
                                 int Mblocks, int Ntiles, int Ktiles, int ldo, int act) {
  int wave = (blockIdx.x * blockDim.x + threadIdx.x) >> 5;
  int lane = threadIdx.x & 31;
  if (wave >= Mblocks * Ntiles) return;     // uniform per wave -> EXEC stays all-ones
  int mb = wave / Ntiles, nt = wave % Ntiles;
  v8f acc[MBLK] = {};
  const size_t astride = (size_t)Ktiles * 512;
  const _Float16* ap = A + (size_t)(mb * MBLK) * astride + lane * 16;
  const _Float16* bp = Bf + (size_t)nt * astride + lane * 16;

  // prologue: load k-tiles 0 and 1 into the two buffer sets
  v16h b0 = *(const v16h*)bp;
  v16h a0[MBLK];
#pragma unroll
  for (int i = 0; i < MBLK; i++) a0[i] = *(const v16h*)(ap + (size_t)i * astride);
  v16h b1 = *(const v16h*)(bp + 512);
  v16h a1[MBLK];
#pragma unroll
  for (int i = 0; i < MBLK; i++) a1[i] = *(const v16h*)(ap + 512 + (size_t)i * astride);

  for (int kt = 0; kt < Ktiles; kt += 2) {
    bool more = (kt + 2) < Ktiles;
    // consume set 0 (k-tile kt)
#pragma unroll
    for (int i = 0; i < MBLK; i++)
      acc[i] = __builtin_amdgcn_wmma_f32_16x16x32_f16(false, a0[i], false, b0, (short)0,
                                                      acc[i], false, false);
    if (more) {                               // reload set 0 for k-tile kt+2
      b0 = *(const v16h*)(bp + 1024);
#pragma unroll
      for (int i = 0; i < MBLK; i++) a0[i] = *(const v16h*)(ap + 1024 + (size_t)i * astride);
    }
    if (kt + 8 < Ktiles) {                    // near-cache prefetch for deep-K streams
      __builtin_prefetch(bp + 8 * 512, 0, 3);
      __builtin_prefetch(ap + 8 * 512, 0, 3);
    }
    // consume set 1 (k-tile kt+1)
#pragma unroll
    for (int i = 0; i < MBLK; i++)
      acc[i] = __builtin_amdgcn_wmma_f32_16x16x32_f16(false, a1[i], false, b1, (short)0,
                                                      acc[i], false, false);
    if (more) {                               // reload set 1 for k-tile kt+3
      b1 = *(const v16h*)(bp + 1024 + 512);
#pragma unroll
      for (int i = 0; i < MBLK; i++) a1[i] = *(const v16h*)(ap + 1536 + (size_t)i * astride);
    }
    ap += 1024; bp += 1024;
  }
  int n = nt * 16 + (lane & 15);
  float bv = bias ? bias[n] : 0.0f;
#pragma unroll
  for (int i = 0; i < MBLK; i++) {
    int mbase = (mb * MBLK + i) * 16 + ((lane >> 4) << 3);
#pragma unroll
    for (int r = 0; r < 8; r++) {
      float vv = acc[i][r] + bv;
      if (act == 1) vv = gelu_exact(vv);
      out[(size_t)(mbase + r) * ldo + n] = vv;
    }
  }
}

// ---------------------------------------------------------------------------
// softmax over rows of 512, one block per 16-row M-tile, emits P fragments
// ---------------------------------------------------------------------------
__global__ void softmax_swz_kernel(const float* __restrict__ S, _Float16* __restrict__ PA) {
  int mt = blockIdx.x;
  __shared__ float P[16][SEA];        // 32 KB
  __shared__ float red[16][16];
  const float* src = S + (size_t)mt * 16 * SEA;
  for (int e = threadIdx.x; e < 16 * SEA; e += 256) P[e >> 9][e & 511] = src[e];
  __syncthreads();
  int r = threadIdx.x >> 4, g = threadIdx.x & 15;   // 16 threads per row
  float mx = -1e30f;
#pragma unroll 4
  for (int j = 0; j < 32; j++) mx = fmaxf(mx, P[r][g * 32 + j]);
  red[r][g] = mx; __syncthreads();
  if (g == 0) { float m = red[r][0]; for (int j = 1; j < 16; j++) m = fmaxf(m, red[r][j]); red[r][0] = m; }
  __syncthreads();
  float rowmax = red[r][0];
  __syncthreads();
  float s = 0.0f;
#pragma unroll 4
  for (int j = 0; j < 32; j++) { float e = __expf(P[r][g * 32 + j] - rowmax); P[r][g * 32 + j] = e; s += e; }
  red[r][g] = s; __syncthreads();
  if (g == 0) { float t = 0; for (int j = 0; j < 16; j++) t += red[r][j]; red[r][0] = t; }
  __syncthreads();
  float inv = 1.0f / red[r][0];
#pragma unroll 4
  for (int j = 0; j < 32; j++) P[r][g * 32 + j] *= inv;
  __syncthreads();
  // emit 16 ktile fragments for this mtile
  for (int pp = threadIdx.x; pp < 16 * 32; pp += 256) {
    int kt = pp >> 5, lane = pp & 31;
    int lr = lane & 15;
    v16h tmp;
#pragma unroll
    for (int h = 0; h < 16; h++) tmp[h] = (_Float16)P[lr][kt * 32 + wmma_k_of(lane, h)];
    *(v16h*)(PA + ((size_t)(mt * 16 + kt) * 512 + lane * 16)) = tmp;
  }
}

// ---------------------------------------------------------------------------
// LayerNorm(128) + exact GELU + residual, emits hb fragments directly
// ---------------------------------------------------------------------------
__global__ void ln_swz_kernel(const float* __restrict__ attn, const float* __restrict__ h32,
                              const float* __restrict__ lnw, const float* __restrict__ lnb,
                              _Float16* __restrict__ hbA) {
  int mt = blockIdx.x;
  __shared__ float hbS[16][DM];       // 8 KB
  __shared__ float r1[16][16], r2[16][16];
  int r = threadIdx.x >> 4, g = threadIdx.x & 15;
  size_t rowbase = ((size_t)mt * 16 + r) * DM;
  float vals[8]; float s = 0.0f, ss = 0.0f;
#pragma unroll
  for (int j = 0; j < 8; j++) { float v = attn[rowbase + g * 8 + j]; vals[j] = v; s += v; ss += v * v; }
  r1[r][g] = s; r2[r][g] = ss; __syncthreads();
  if (g == 0) { float a = 0, b2 = 0; for (int j = 0; j < 16; j++) { a += r1[r][j]; b2 += r2[r][j]; } r1[r][0] = a; r2[r][0] = b2; }
  __syncthreads();
  float mu = r1[r][0] / (float)DM;
  float var = r2[r][0] / (float)DM - mu * mu;
  float rstd = rsqrtf(var + EPS);
#pragma unroll
  for (int j = 0; j < 8; j++) {
    int c = g * 8 + j;
    float ln = (vals[j] - mu) * rstd * lnw[c] + lnb[c];
    hbS[r][c] = gelu_exact(ln) + h32[rowbase + c];
  }
  __syncthreads();
  if (threadIdx.x < 128) {            // 4 ktiles x 32 lanes
    int kt = threadIdx.x >> 5, lane = threadIdx.x & 31;
    int lr = lane & 15;
    v16h tmp;
#pragma unroll
    for (int h = 0; h < 16; h++) tmp[h] = (_Float16)hbS[lr][kt * 32 + wmma_k_of(lane, h)];
    *(v16h*)(hbA + ((size_t)(mt * 4 + kt) * 512 + lane * 16)) = tmp;
  }
}

// ---------------------------------------------------------------------------
// Mamba core: conv + silu + x_proj + dt_proj/softplus + scan + D-skip +
// silu(z) gate; gated y written back into the xc LDS tile, then emitted as
// A fragments (ygA) directly -> no ygate global buffer, no extra swizzle pass.
// ---------------------------------------------------------------------------
__global__ void mamba_kernel(const float* __restrict__ xz, const float* __restrict__ conv_w,
                             const float* __restrict__ conv_b, const float* __restrict__ xproj_w,
                             const float* __restrict__ dtw, const float* __restrict__ dtb,
                             const float* __restrict__ A_log, const float* __restrict__ Dssm,
                             _Float16* __restrict__ ygA) {
  int seq = blockIdx.x;
  int c = threadIdx.x;                       // 0..255 (= DIN)
  __shared__ float xc[PN][DIN];              // 64 KB (later holds gated y)
  __shared__ float dbl[PN][40];              // 10 KB
  const float* xzrow = xz + (size_t)seq * PN * 512;

  // causal depthwise conv (K=4) + bias + silu along L, channel c
  float w0 = conv_w[c * DC + 0], w1 = conv_w[c * DC + 1];
  float w2 = conv_w[c * DC + 2], w3 = conv_w[c * DC + 3];
  float cb = conv_b[c];
  float xm3 = 0.0f, xm2 = 0.0f, xm1 = 0.0f;
  for (int l = 0; l < PN; l++) {
    float x0 = xzrow[l * 512 + c];
    xc[l][c] = silu_f(w0 * xm3 + w1 * xm2 + w2 * xm1 + w3 * x0 + cb);
    xm3 = xm2; xm2 = xm1; xm1 = x0;
  }
  __syncthreads();

  // x_proj: (64 x 256) @ (256 x 40) -> dbl; 10 outputs per thread
  for (int o = c * 10; o < c * 10 + 10; o++) {
    int l = o / 40, j = o % 40;
    const float* wr = xproj_w + j * DIN;
    float acc = 0.0f;
    for (int k = 0; k < DIN; k++) acc += xc[l][k] * wr[k];
    dbl[l][j] = acc;
  }
  __syncthreads();

  // selective scan, channel c owns a (DS=16)-wide state
  float Ac[DS], dw[DTR], hst[DS];
#pragma unroll
  for (int s = 0; s < DS; s++) { Ac[s] = -__expf(A_log[c * DS + s]); hst[s] = 0.0f; }
#pragma unroll
  for (int j = 0; j < DTR; j++) dw[j] = dtw[c * DTR + j];
  float db = dtb[c], Dc = Dssm[c];

  for (int l = 0; l < PN; l++) {
    float dt = db;
#pragma unroll
    for (int j = 0; j < DTR; j++) dt += dbl[l][j] * dw[j];
    float d = softplus_f(dt);
    float u = xc[l][c];                     // column c only: no cross-thread race
    float du = d * u;
    float y = 0.0f;
#pragma unroll
    for (int s = 0; s < DS; s++) {
      float dA = __expf(d * Ac[s]);
      hst[s] = dA * hst[s] + du * dbl[l][8 + s];
      y += hst[s] * dbl[l][24 + s];
    }
    y += Dc * u;
    float z = xzrow[l * 512 + 256 + c];
    xc[l][c] = y * silu_f(z);               // overwrite in place with gated output
  }
  __syncthreads();

  // emit A fragments: 4 local mtiles x 8 ktiles (K = 256)
  for (int pp = threadIdx.x; pp < 32 * 32; pp += blockDim.x) {
    int lt = pp >> 5, lane = pp & 31;
    int lmt = lt >> 3, lkt = lt & 7;
    int lr = lmt * 16 + (lane & 15);
    v16h tmp;
#pragma unroll
    for (int h = 0; h < 16; h++) tmp[h] = (_Float16)xc[lr][lkt * 32 + wmma_k_of(lane, h)];
    size_t tile = (size_t)(seq * 4 + lmt) * 8 + lkt;
    *(v16h*)(ygA + tile * 512 + lane * 16) = tmp;
  }
}

// ---------------------------------------------------------------------------
// mlp3 (192->96) + inverse RevIN + transpose to (B, PRED, NV)
// ---------------------------------------------------------------------------
__global__ void mlp3_final_kernel(const float* __restrict__ g, const float* __restrict__ w3,
                                  const float* __restrict__ b3, const float* __restrict__ revin_w,
                                  const float* __restrict__ revin_b, const float* __restrict__ mean,
                                  const float* __restrict__ stdv, float* __restrict__ out) {
  int seq = blockIdx.x; int b = seq / NV, v = seq % NV;
  int p = threadIdx.x;            // 0..95
  const float* gr = g + (size_t)seq * 192;
  const float* wr = w3 + (size_t)p * 192;
  float acc = b3[p];
  for (int j = 0; j < 192; j++) acc += gr[j] * wr[j];
  float o2 = (acc - revin_b[v]) / (revin_w[v] + 1e-10f);   // EPS*EPS
  out[((size_t)b * PRED + p) * NV + v] = o2 * stdv[seq] + mean[seq];
}

// ---------------------------------------------------------------------------
// launcher
// ---------------------------------------------------------------------------
extern "C" void kernel_launch(void* const* d_in, const int* in_sizes, int n_in,
                              void* d_out, int out_size, void* d_ws, size_t ws_size,
                              hipStream_t stream) {
  (void)in_sizes; (void)n_in; (void)out_size; (void)ws_size;
  const float* x         = (const float*)d_in[0];
  const float* revin_w   = (const float*)d_in[1];
  const float* revin_b   = (const float*)d_in[2];
  const float* mlp1_w    = (const float*)d_in[3];
  const float* mlp1_b    = (const float*)d_in[4];
  const float* mk_w      = (const float*)d_in[5];
  const float* mv_w      = (const float*)d_in[6];
  const float* ln_w      = (const float*)d_in[7];
  const float* ln_b      = (const float*)d_in[8];
  const float* in_proj_w = (const float*)d_in[9];
  const float* conv_w    = (const float*)d_in[10];
  const float* conv_b    = (const float*)d_in[11];
  const float* x_proj_w  = (const float*)d_in[12];
  const float* dt_proj_w = (const float*)d_in[13];
  const float* dt_proj_b = (const float*)d_in[14];
  const float* A_log     = (const float*)d_in[15];
  const float* D_ssm     = (const float*)d_in[16];
  const float* out_proj_w= (const float*)d_in[17];
  const float* mlp2_w    = (const float*)d_in[18];
  const float* mlp2_b    = (const float*)d_in[19];
  const float* mlp3_w    = (const float*)d_in[20];
  const float* mlp3_b    = (const float*)d_in[21];
  float* out = (float*)d_out;

  // ---- workspace carve-up (regions reused across pipeline stages) ----
  char* base = (char*)d_ws;
  size_t off = 0;
  auto alloc = [&](size_t bytes) -> char* {
    char* p = base + off;
    off = (off + bytes + 255) & ~(size_t)255;
    return p;
  };
  float* mean   = (float*)alloc(BN * 4);
  float* stdv   = (float*)alloc(BN * 4);
  float* h32    = (float*)alloc((size_t)ROWS * DM * 4);
  float* S32    = (float*)alloc((size_t)ROWS * SEA * 4);   // later reused as xz
  float* attn32 = (float*)alloc((size_t)ROWS * DM * 4);    // later reused as y_mamba
  float* g32    = (float*)alloc((size_t)BN * 192 * 4);
  _Float16* hA  = (_Float16*)alloc((size_t)ROWS * DM * 2);
  _Float16* hbA = (_Float16*)alloc((size_t)ROWS * DM * 2);
  _Float16* PA  = (_Float16*)alloc((size_t)ROWS * SEA * 2); // reused for ygA / yA
  _Float16* mkB = (_Float16*)alloc((size_t)SEA * DM * 2);
  _Float16* mvB = (_Float16*)alloc((size_t)SEA * DM * 2);
  _Float16* ipB = (_Float16*)alloc((size_t)512 * DM * 2);
  _Float16* opB = (_Float16*)alloc((size_t)DIN * DM * 2);
  _Float16* m2B = (_Float16*)alloc((size_t)192 * 8192 * 2);
  float* xz32 = S32;                  // S dead after softmax_swz
  float* ymam = attn32;               // attn dead after ln_swz
  _Float16* ygA = PA;                 // P-frags dead after attn gemm
  _Float16* yA  = PA + (size_t)ROWS * DIN;

  auto swzB = [&](const float* W, _Float16* o, int N, int K) {
    int nt = N / 16, kt = K / 32, thr = nt * kt * 32;
    swizzleB_kernel<<<(thr + 255) / 256, 256, 0, stream>>>(W, o, nt, kt, K);
  };
  auto gemm4 = [&](const _Float16* A, const _Float16* Bf, const float* bias, float* o,
                   int Mt, int Nt, int Kt, int ldo, int act) {
    int mb = Mt / 4;
    long waves = (long)mb * Nt;
    int blocks = (int)((waves * 32 + 255) / 256);
    wmma_gemm_kernel<4><<<blocks, 256, 0, stream>>>(A, Bf, bias, o, mb, Nt, Kt, ldo, act);
  };
  auto gemm3 = [&](const _Float16* A, const _Float16* Bf, const float* bias, float* o,
                   int Mt, int Nt, int Kt, int ldo, int act) {
    int mb = Mt / 3;
    long waves = (long)mb * Nt;
    int blocks = (int)((waves * 32 + 255) / 256);
    wmma_gemm_kernel<3><<<blocks, 256, 0, stream>>>(A, Bf, bias, o, mb, Nt, Kt, ldo, act);
  };

  // weight fragment conversion (tiny, once per call)
  swzB(mk_w,       mkB, 512, 128);
  swzB(mv_w,       mvB, 128, 512);
  swzB(in_proj_w,  ipB, 512, 128);
  swzB(out_proj_w, opB, 128, 256);
  swzB(mlp2_w,     m2B, 192, 8192);

  // 1-2: stats, patchify+mlp1 (emits h32 + hA)
  stats_kernel<<<BN, 256, 0, stream>>>(x, mean, stdv);
  patch_mlp1_kernel<<<BN, 256, 0, stream>>>(x, revin_w, revin_b, mlp1_w, mlp1_b,
                                            mean, stdv, h32, hA);

  // 3: external attention  S = h @ mk^T ; fused softmax+swizzle ; attn = P @ mv^T
  gemm4(hA, mkB, nullptr, S32, ROWS / 16, SEA / 16, DM / 32, SEA, 0);
  softmax_swz_kernel<<<ROWS / 16, 256, 0, stream>>>(S32, PA);
  gemm4(PA, mvB, nullptr, attn32, ROWS / 16, DM / 16, SEA / 32, DM, 0);

  // 4: LN + gelu + residual -> hbA fragments (no hb32 round trip)
  ln_swz_kernel<<<ROWS / 16, 256, 0, stream>>>(attn32, h32, ln_w, ln_b, hbA);

  // 5: in_proj  xz = hb @ in_proj^T  (cols 0..255 = xc_raw, 256..511 = z)
  gemm4(hbA, ipB, nullptr, xz32, ROWS / 16, 512 / 16, DM / 32, 512, 0);

  // 6: mamba core -> ygA fragments directly
  mamba_kernel<<<BN, 256, 0, stream>>>(xz32, conv_w, conv_b, x_proj_w, dt_proj_w,
                                       dt_proj_b, A_log, D_ssm, ygA);

  // 7: out_proj  y = ygate @ out_proj^T
  gemm4(ygA, opB, nullptr, ymam, ROWS / 16, DM / 16, DIN / 32, DM, 0);

  // 8: mlp2 over the (336 x 8192) view + gelu  (Mtiles=21 -> MBLK=3)
  {
    int mt = BN / 16, kt = (PN * DM) / 32, thr = mt * kt * 32;
    swizzleA_kernel<<<(thr + 255) / 256, 256, 0, stream>>>(ymam, yA, mt, kt, PN * DM);
  }
  gemm3(yA, m2B, mlp2_b, g32, BN / 16, 192 / 16, (PN * DM) / 32, 192, 1);

  // 9: mlp3 + inverse RevIN + transpose
  mlp3_final_kernel<<<BN, PRED, 0, stream>>>(g32, mlp3_w, mlp3_b, revin_w, revin_b,
                                             mean, stdv, out);
}
